// SafeGNN_49632642073138
// MI455X (gfx1250) — matrix operand order repaired and checked by reading, not compiled, not run on hardware
//
#include <hip/hip_runtime.h>
#include <hip/hip_bf16.h>

#define N_NODES   100000
#define N_EDGES   1600000
#define N_GRAPHS  64
#define HDIM      64
#define NLAYERS   3
#define EDGE_TILES (N_EDGES / 16)   // 100000
#define NODE_TILES (N_NODES / 16)   // 6250

typedef __attribute__((ext_vector_type(16))) __bf16 v16bf;
typedef __attribute__((ext_vector_type(8)))  float  v8f;
typedef __attribute__((ext_vector_type(4)))  unsigned int v4u;  // native vec for b128 loads

union V16 { v16bf v; v4u q[2]; };   // direct register-pair assembly, no elementwise moves

__device__ __forceinline__ float sigmoid_f(float x) { return 1.0f / (1.0f + __expf(-x)); }

// ---------------------------------------------------------------------------
// Utility: zero a float buffer
// ---------------------------------------------------------------------------
__global__ void zero_f32_kernel(float* __restrict__ p, long n) {
  long i = (long)blockIdx.x * blockDim.x + threadIdx.x;
  if (i < n) p[i] = 0.0f;
}

// ---------------------------------------------------------------------------
// Pack 9 64x64 fp32 weight matrices into per-lane WMMA B-operand bf16 layout.
// Layout per matrix: [t(0..3)][kchunk(0..1)][lane(0..31)][i(0..15)]
//   element i of the v16bf in lane L is B[K][N] with
//   K = kchunk*32 + (L<16 ? 0 : 8) + (i<8 ? i : 16 + (i-8)),  N = t*16 + (L&15)
// Matrices 0..2 = lin_e_w[l], 3..5 = mlp_w1[l], 6..8 = mlp_w2[l]
// ---------------------------------------------------------------------------
__global__ void pack_weights_kernel(const float* __restrict__ lin_e_w,
                                    const float* __restrict__ mlp_w1,
                                    const float* __restrict__ mlp_w2,
                                    __bf16* __restrict__ Bp) {
  int idx = blockIdx.x * blockDim.x + threadIdx.x;
  if (idx >= 9 * 4096) return;
  int mat  = idx / 4096;
  int r    = idx & 4095;
  int i    = r & 15;
  int lane = (r >> 4) & 31;
  int kc   = (r >> 9) & 1;
  int t    = (r >> 10) & 3;
  int K    = kc * 32 + ((lane < 16) ? 0 : 8) + ((i < 8) ? i : 16 + (i - 8));
  int col  = t * 16 + (lane & 15);
  const float* W;
  int l;
  if (mat < 3)      { W = lin_e_w; l = mat;     }
  else if (mat < 6) { W = mlp_w1;  l = mat - 3; }
  else              { W = mlp_w2;  l = mat - 6; }
  Bp[idx] = (__bf16)W[(size_t)l * 4096 + K * 64 + col];
}

// ---------------------------------------------------------------------------
// Node encoder: h = x @ W(3x64) + b     (fp32, memory-trivial)
// ---------------------------------------------------------------------------
__global__ void node_encoder_kernel(const float* __restrict__ x,
                                    const float* __restrict__ W,
                                    const float* __restrict__ b,
                                    float* __restrict__ h) {
  int i = blockIdx.x * blockDim.x + threadIdx.x;
  if (i >= N_NODES * HDIM) return;
  int nn = i >> 6, col = i & 63;
  const float* xr = x + nn * 3;
  h[i] = xr[0] * W[col] + xr[1] * W[64 + col] + xr[2] * W[128 + col] + b[col];
}

// ---------------------------------------------------------------------------
// Edge encoder: ea = bf16(edge_attr @ W(2x64) + b)   (streamed NT store)
// ---------------------------------------------------------------------------
__global__ void edge_encoder_kernel(const float* __restrict__ ea_in,
                                    const float* __restrict__ W,
                                    const float* __restrict__ b,
                                    __bf16* __restrict__ ea_out) {
  long i = (long)blockIdx.x * blockDim.x + threadIdx.x;
  if (i >= (long)N_EDGES * HDIM) return;
  long e = i >> 6; int col = (int)(i & 63);
  float v = ea_in[e * 2] * W[col] + ea_in[e * 2 + 1] * W[64 + col] + b[col];
  __builtin_nontemporal_store((__bf16)v, ea_out + i);
}

// ---------------------------------------------------------------------------
// Fused edge layer: ea_l = ea @ lin_e_w + b (WMMA bf16->f32),
// msg = relu(h[src] + ea_l), atomic scatter-add into aggr[dst].
// One wave per 16-edge tile (stride loop). blockDim = 256 (8 waves).
// B operands staged in LDS (8KB/block) to keep VGPR pressure low (no spills).
// ---------------------------------------------------------------------------
__global__ void edge_layer_kernel(const __bf16* __restrict__ ea,   // [E,64]
                                  const __bf16* __restrict__ Bp,   // packed lin_e_w[l]
                                  const float*  __restrict__ bias, // lin_e_b[l], 64
                                  const int*    __restrict__ src,
                                  const int*    __restrict__ dst,
                                  const float*  __restrict__ h,    // [N,64]
                                  float*        __restrict__ aggr) // [N,64]
{
  __shared__ __bf16 bsh[4096];   // packed B, 8 KB

  const int lane  = threadIdx.x & 31;
  const int wave  = threadIdx.x >> 5;
  const int gwave = blockIdx.x * (blockDim.x >> 5) + wave;
  const int nwave = gridDim.x * (blockDim.x >> 5);

  const int n    = lane & 15;
  const int moff = (lane < 16) ? 0 : 8;  // D-rows this lane-half owns
  const int b0a  = (lane < 16) ? 0 : 8;  // A K-offset for this lane-half

  // cooperative stage of packed B into LDS
  {
    v4u* d = (v4u*)bsh;
    const v4u* s = (const v4u*)Bp;
    for (int i = threadIdx.x; i < 512; i += blockDim.x) d[i] = s[i];
  }
  __syncthreads();

  float bias_t[4];
#pragma unroll
  for (int t = 0; t < 4; t++) bias_t[t] = bias[t * 16 + n];

  for (int tile = gwave; tile < EDGE_TILES; tile += nwave) {
    asm volatile("" ::: "memory");  // keep LDS B loads inside the loop (no LICM re-hoist)
    const int e0 = tile * 16;

    // A operand: 16 edges x 64 bf16, ISA 16-bit A layout, NT streaming loads
    V16 A[2];
    const __bf16* rowp = ea + (size_t)(e0 + n) * HDIM;
#pragma unroll
    for (int kc = 0; kc < 2; kc++) {
      const __bf16* p = rowp + kc * 32 + b0a;
      A[kc].q[0] = __builtin_nontemporal_load((const v4u*)(p));
      A[kc].q[1] = __builtin_nontemporal_load((const v4u*)(p + 16));
    }

    int se[8], de[8];
#pragma unroll
    for (int j = 0; j < 8; j++) {
      se[j] = src[e0 + moff + j];
      de[j] = dst[e0 + moff + j];
    }

#pragma unroll
    for (int t = 0; t < 4; t++) {
      // B operands from LDS (cheap, keeps registers free)
      V16 B0, B1;
      {
        const __bf16* pb0 = bsh + ((t * 2 + 0) * 32 + lane) * 16;
        const __bf16* pb1 = bsh + ((t * 2 + 1) * 32 + lane) * 16;
        B0.q[0] = *(const v4u*)pb0;  B0.q[1] = *(const v4u*)(pb0 + 8);
        B1.q[0] = *(const v4u*)pb1;  B1.q[1] = *(const v4u*)(pb1 + 8);
      }
      v8f acc = {};
      acc = __builtin_amdgcn_wmma_f32_16x16x32_bf16(false, A[0].v, false, B0.v,
                                                    (short)0, acc, false, false);
      acc = __builtin_amdgcn_wmma_f32_16x16x32_bf16(false, A[1].v, false, B1.v,
                                                    (short)0, acc, false, false);
      const unsigned col = (unsigned)(t * 16 + n);

      // batch the 8 gathers (SADDR + 32-bit voffset form), then the 8 atomics
      float hv[8];
#pragma unroll
      for (int j = 0; j < 8; j++)
        hv[j] = h[(unsigned)se[j] * 64u + col];        // L2-resident gather
#pragma unroll
      for (int j = 0; j < 8; j++) {
        float v = acc[j] + bias_t[t] + hv[j];
        v = v > 0.0f ? v : 0.0f;                        // ReLU
        atomicAdd(&aggr[(unsigned)de[j] * 64u + col], v); // L2 f32 atomic
      }
    }
  }
}

// ---------------------------------------------------------------------------
// Fused node layer: z = (1+eps)h + aggr ; z = silu(z@W1+b1)@W2+b2 ;
// h = silu(z) + h.   Two WMMA GEMMs, LDS bounce to re-enter A layout.
// One wave per 16-node tile. blockDim = 256. W1/W2 packs staged in LDS.
// ---------------------------------------------------------------------------
__global__ void node_layer_kernel(float* __restrict__ h,
                                  const float*  __restrict__ aggr,
                                  const __bf16* __restrict__ W1p,
                                  const float*  __restrict__ b1,
                                  const __bf16* __restrict__ W2p,
                                  const float*  __restrict__ b2,
                                  const float*  __restrict__ epsp)
{
  __shared__ __bf16 b1sh[4096];        // 8 KB
  __shared__ __bf16 b2sh[4096];        // 8 KB
  __shared__ __bf16 smem[8 * 16 * HDIM];  // 16 KB activation bounce

  const int lane  = threadIdx.x & 31;
  const int wave  = threadIdx.x >> 5;
  const int gwave = blockIdx.x * (blockDim.x >> 5) + wave;
  const int nwave = gridDim.x * (blockDim.x >> 5);

  const int n    = lane & 15;
  const int moff = (lane < 16) ? 0 : 8;
  const int b0a  = (lane < 16) ? 0 : 8;
  __bf16* lds = smem + wave * (16 * HDIM);

  // cooperative stage of both weight packs into LDS
  {
    v4u* d1 = (v4u*)b1sh; const v4u* s1 = (const v4u*)W1p;
    v4u* d2 = (v4u*)b2sh; const v4u* s2 = (const v4u*)W2p;
    for (int i = threadIdx.x; i < 512; i += blockDim.x) { d1[i] = s1[i]; d2[i] = s2[i]; }
  }
  __syncthreads();

  const float epsv = 1.0f + *epsp;
  float b1v[4], b2v[4];
#pragma unroll
  for (int t = 0; t < 4; t++) { b1v[t] = b1[t * 16 + n]; b2v[t] = b2[t * 16 + n]; }

  for (int tile = gwave; tile < NODE_TILES; tile += nwave) {
    asm volatile("" ::: "memory");  // keep LDS B loads inside the loop
    const int n0 = tile * 16;

    // zin = (1+eps)*h + aggr, built directly in bf16 A layout
    V16 A[2];
#pragma unroll
    for (int kc = 0; kc < 2; kc++) {
      const unsigned rbase = (unsigned)(n0 + n) * 64u + (unsigned)(kc * 32 + b0a);
#pragma unroll
      for (int i = 0; i < 8; i++) {
        A[kc].v[i]     = (__bf16)(epsv * h[rbase + i]      + aggr[rbase + i]);
        A[kc].v[8 + i] = (__bf16)(epsv * h[rbase + 16 + i] + aggr[rbase + 16 + i]);
      }
    }

    // GEMM1 + SiLU -> LDS (bf16, D layout written row/col-wise)
#pragma unroll
    for (int t = 0; t < 4; t++) {
      V16 B0, B1;
      const __bf16* pb0 = b1sh + ((t * 2 + 0) * 32 + lane) * 16;
      const __bf16* pb1 = b1sh + ((t * 2 + 1) * 32 + lane) * 16;
      B0.q[0] = *(const v4u*)pb0;  B0.q[1] = *(const v4u*)(pb0 + 8);
      B1.q[0] = *(const v4u*)pb1;  B1.q[1] = *(const v4u*)(pb1 + 8);
      v8f acc = {};
      acc = __builtin_amdgcn_wmma_f32_16x16x32_bf16(false, A[0].v, false, B0.v,
                                                    (short)0, acc, false, false);
      acc = __builtin_amdgcn_wmma_f32_16x16x32_bf16(false, A[1].v, false, B1.v,
                                                    (short)0, acc, false, false);
      const int col = t * 16 + n;
#pragma unroll
      for (int j = 0; j < 8; j++) {
        float v = acc[j] + b1v[t];
        v = v * sigmoid_f(v);                          // SiLU
        lds[(moff + j) * HDIM + col] = (__bf16)v;
      }
    }

    // same-wave LDS ordering: DS ops are in-order, drain DScnt + compiler fence
    asm volatile("s_wait_dscnt 0" ::: "memory");

    // reload activations in A layout from LDS
    V16 A2[2];
#pragma unroll
    for (int kc = 0; kc < 2; kc++) {
      const __bf16* p = lds + n * HDIM + kc * 32 + b0a;
      A2[kc].q[0] = *(const v4u*)(p);
      A2[kc].q[1] = *(const v4u*)(p + 16);
    }

    // GEMM2 + bias + SiLU + residual, in-place update of h (tile-exclusive rows)
#pragma unroll
    for (int t = 0; t < 4; t++) {
      V16 B0, B1;
      const __bf16* pb0 = b2sh + ((t * 2 + 0) * 32 + lane) * 16;
      const __bf16* pb1 = b2sh + ((t * 2 + 1) * 32 + lane) * 16;
      B0.q[0] = *(const v4u*)pb0;  B0.q[1] = *(const v4u*)(pb0 + 8);
      B1.q[0] = *(const v4u*)pb1;  B1.q[1] = *(const v4u*)(pb1 + 8);
      v8f acc = {};
      acc = __builtin_amdgcn_wmma_f32_16x16x32_bf16(false, A2[0].v, false, B0.v,
                                                    (short)0, acc, false, false);
      acc = __builtin_amdgcn_wmma_f32_16x16x32_bf16(false, A2[1].v, false, B1.v,
                                                    (short)0, acc, false, false);
      const unsigned col = (unsigned)(t * 16 + n);
#pragma unroll
      for (int j = 0; j < 8; j++) {
        unsigned idx = (unsigned)(n0 + moff + j) * 64u + col;
        float z2 = acc[j] + b2v[t];
        h[idx] = z2 * sigmoid_f(z2) + h[idx];
      }
    }
  }
}

// ---------------------------------------------------------------------------
// Global mean pool accumulation (L2 atomics; gsum/counts tiny)
// ---------------------------------------------------------------------------
__global__ void pool_kernel(const float* __restrict__ h,
                            const int* __restrict__ batch,
                            float* __restrict__ gsum,
                            float* __restrict__ counts) {
  int i = blockIdx.x * blockDim.x + threadIdx.x;
  if (i >= N_NODES * HDIM) return;
  int nn = i >> 6, col = i & 63;
  int g = batch[nn];
  atomicAdd(&gsum[g * HDIM + col], h[i]);
  if (col == 0) atomicAdd(&counts[g], 1.0f);
}

// ---------------------------------------------------------------------------
// Head MLP: feat=[dc_h, mean_h] (128) -> 128 -> 64 -> 1 -> softplus.
// One block per graph (64 blocks x 128 threads). Trivial FLOPs.
// ---------------------------------------------------------------------------
__global__ void head_kernel(const float* __restrict__ h,
                            const float* __restrict__ gsum,
                            const float* __restrict__ counts,
                            const int*   __restrict__ dc_mask,
                            const float* __restrict__ W1, const float* __restrict__ bb1,
                            const float* __restrict__ W2, const float* __restrict__ bb2,
                            const float* __restrict__ W3, const float* __restrict__ bb3,
                            float* __restrict__ out) {
  __shared__ float feat[128];
  __shared__ float o1[128];
  __shared__ float o2[64];
  int g = blockIdx.x, t = threadIdx.x;
  if (t < 64) {
    feat[t] = h[(size_t)dc_mask[g] * HDIM + t];
  } else {
    float c = fmaxf(counts[g], 1.0f);
    feat[t] = gsum[g * HDIM + (t - 64)] / c;
  }
  __syncthreads();
  float a1 = bb1[t];
#pragma unroll 4
  for (int k = 0; k < 128; k++) a1 += feat[k] * W1[k * 128 + t];
  o1[t] = a1 * sigmoid_f(a1);
  __syncthreads();
  if (t < 64) {
    float a2 = bb2[t];
#pragma unroll 4
    for (int k = 0; k < 128; k++) a2 += o1[k] * W2[k * 64 + t];
    o2[t] = a2 * sigmoid_f(a2);
  }
  __syncthreads();
  if (t == 0) {
    float a3 = bb3[0];
#pragma unroll 4
    for (int k = 0; k < 64; k++) a3 += o2[k] * W3[k];
    // numerically stable softplus
    out[g] = fmaxf(a3, 0.0f) + log1pf(__expf(-fabsf(a3)));
  }
}

// ---------------------------------------------------------------------------
// Orchestration
// ---------------------------------------------------------------------------
extern "C" void kernel_launch(void* const* d_in, const int* in_sizes, int n_in,
                              void* d_out, int out_size, void* d_ws, size_t ws_size,
                              hipStream_t stream) {
  const float* x          = (const float*)d_in[0];
  const int*   edge_index = (const int*)  d_in[1];   // [2,E]
  const float* edge_attr  = (const float*)d_in[2];
  const int*   batch      = (const int*)  d_in[3];
  const int*   dc_mask    = (const int*)  d_in[4];
  const float* node_enc_w = (const float*)d_in[5];
  const float* node_enc_b = (const float*)d_in[6];
  const float* edge_enc_w = (const float*)d_in[7];
  const float* edge_enc_b = (const float*)d_in[8];
  const float* lin_e_w    = (const float*)d_in[9];
  const float* lin_e_b    = (const float*)d_in[10];
  const float* mlp_w1     = (const float*)d_in[11];
  const float* mlp_b1     = (const float*)d_in[12];
  const float* mlp_w2     = (const float*)d_in[13];
  const float* mlp_b2     = (const float*)d_in[14];
  const float* eps        = (const float*)d_in[15];
  const float* head_w1    = (const float*)d_in[16];
  const float* head_b1    = (const float*)d_in[17];
  const float* head_w2    = (const float*)d_in[18];
  const float* head_b2    = (const float*)d_in[19];
  const float* head_w3    = (const float*)d_in[20];
  const float* head_b3    = (const float*)d_in[21];
  float* out = (float*)d_out;

  // workspace carve (256B aligned)
  size_t off = 0;
  auto carve = [&](size_t bytes) -> char* {
    char* p = (char*)d_ws + off;
    off += (bytes + 255) & ~(size_t)255;
    return p;
  };
  __bf16* ea   = (__bf16*)carve((size_t)N_EDGES * HDIM * 2);   // 204.8 MB
  float*  h    = (float*) carve((size_t)N_NODES * HDIM * 4);   // 25.6 MB
  float*  aggr = (float*) carve((size_t)N_NODES * HDIM * 4);   // 25.6 MB
  __bf16* Bp   = (__bf16*)carve((size_t)9 * 4096 * 2);         // 72 KB
  float*  gsum = (float*) carve((size_t)N_GRAPHS * HDIM * 4);
  float*  cnts = (float*) carve((size_t)N_GRAPHS * 4);
  (void)ws_size; (void)in_sizes; (void)n_in; (void)out_size;

  const int* src = edge_index;
  const int* dst = edge_index + N_EDGES;

  // weight packing + encoders
  pack_weights_kernel<<<(9 * 4096 + 255) / 256, 256, 0, stream>>>(lin_e_w, mlp_w1, mlp_w2, Bp);
  node_encoder_kernel<<<(N_NODES * HDIM + 255) / 256, 256, 0, stream>>>(x, node_enc_w, node_enc_b, h);
  edge_encoder_kernel<<<(int)(((long)N_EDGES * HDIM + 255) / 256), 256, 0, stream>>>(
      edge_attr, edge_enc_w, edge_enc_b, ea);

  // GINE layers
  for (int l = 0; l < NLAYERS; l++) {
    zero_f32_kernel<<<(N_NODES * HDIM + 255) / 256, 256, 0, stream>>>(aggr, (long)N_NODES * HDIM);
    edge_layer_kernel<<<2048, 256, 0, stream>>>(ea, Bp + (size_t)l * 4096,
                                                lin_e_b + l * HDIM, src, dst, h, aggr);
    node_layer_kernel<<<512, 256, 0, stream>>>(h, aggr,
                                               Bp + (size_t)(3 + l) * 4096, mlp_b1 + l * HDIM,
                                               Bp + (size_t)(6 + l) * 4096, mlp_b2 + l * HDIM,
                                               eps + l);
  }

  // pooling + head
  zero_f32_kernel<<<(N_GRAPHS * HDIM + 255) / 256, 256, 0, stream>>>(gsum, (long)N_GRAPHS * HDIM);
  zero_f32_kernel<<<1, 64, 0, stream>>>(cnts, (long)N_GRAPHS);
  pool_kernel<<<(N_NODES * HDIM + 255) / 256, 256, 0, stream>>>(h, batch, gsum, cnts);
  head_kernel<<<N_GRAPHS, 128, 0, stream>>>(h, gsum, cnts, dc_mask,
                                            head_w1, head_b1, head_w2, head_b2,
                                            head_w3, head_b3, out);
}
